// MainModelAttention_31147102830913
// MI455X (gfx1250) — compile-verified, matrix operand order checked by reading, not run on hardware
//
#include <hip/hip_runtime.h>
#include <hip/hip_bf16.h>

// ---------------------------------------------------------------------------
// Types for CDNA5 WMMA (wave32): bf16 A/B fragments (16 x __bf16 = 8 VGPRs),
// f32 accumulator (8 floats = 8 VGPRs).
// ---------------------------------------------------------------------------
typedef __bf16 bf16_t;
typedef bf16_t v16bf __attribute__((ext_vector_type(16)));
typedef float  v8f   __attribute__((ext_vector_type(8)));

typedef unsigned int u32x4v __attribute__((ext_vector_type(4)));
typedef int          i32x4v __attribute__((ext_vector_type(4)));
typedef int          i32x8v __attribute__((ext_vector_type(8)));

union BF16Frag {
    v16bf v;
    uint4 q[2];
    unsigned short h[16];
};

union Pack8 {
    unsigned short h[8];
    uint4 q;
};

__device__ __forceinline__ unsigned short f32_to_bf16(float f) {
    union { float f; unsigned int u; } a;
    a.f = f;
    unsigned int u = a.u;
    unsigned int r = u + 0x7FFFu + ((u >> 16) & 1u);   // round-to-nearest-even
    return (unsigned short)(r >> 16);
}

#define H_DIM   768
#define S_LEN   4096
#define NTOK    16384          // B * S

// TDM issue: one 2D bf16 tile (32 rows x 768 cols, row stride 768) -> LDS.
__device__ __forceinline__ void tdm_load_k32(unsigned lds_addr,
                                             const unsigned short* gptr,
                                             const i32x8v& g1)
{
    const unsigned long long ga = (unsigned long long)(uintptr_t)gptr;
    u32x4v g0;
    g0[0] = 1u;                                   // count=1, no gather
    g0[1] = lds_addr;                             // LDS byte address
    g0[2] = (unsigned)ga;                         // global_addr lo
    g0[3] = (unsigned)((ga >> 32) & 0x01FFFFFFu)  // global_addr[56:32]
          | (2u << 30);                           // D# type = 2 ("image")
    const i32x4v gz4 = {0, 0, 0, 0};
#if defined(__clang_major__) && (__clang_major__ >= 23)
    const i32x8v gz8 = {0, 0, 0, 0, 0, 0, 0, 0};
    __builtin_amdgcn_tensor_load_to_lds(g0, g1, gz4, gz4, gz8, 0);
#else
    __builtin_amdgcn_tensor_load_to_lds(g0, g1, gz4, gz4, 0);
#endif
}

// ---------------------------------------------------------------------------
// Kernel 0: bulk fp32 -> bf16 conversion (memory-bound, one pass).
// ---------------------------------------------------------------------------
__global__ __launch_bounds__(256) void cvt_f32_bf16_kernel(
    const float* __restrict__ src, unsigned short* __restrict__ dst, int n8)
{
    const int i = blockIdx.x * blockDim.x + threadIdx.x;
    if (i >= n8) return;
    const float4* s = (const float4*)src + (size_t)i * 2;
    const float4 a = s[0];
    const float4 b = s[1];
    Pack8 pk;
    pk.h[0] = f32_to_bf16(a.x); pk.h[1] = f32_to_bf16(a.y);
    pk.h[2] = f32_to_bf16(a.z); pk.h[3] = f32_to_bf16(a.w);
    pk.h[4] = f32_to_bf16(b.x); pk.h[5] = f32_to_bf16(b.y);
    pk.h[6] = f32_to_bf16(b.z); pk.h[7] = f32_to_bf16(b.w);
    ((uint4*)dst)[i] = pk.q;
}

// ---------------------------------------------------------------------------
// Kernel 1: fused QKV projection (all-bf16 inputs).
// Each wave computes a 16x96 strip (6 accum tiles); B fragments are
// preloaded as a batch so the loads clause together and waits are partial.
// ---------------------------------------------------------------------------
__global__ __launch_bounds__(256) void qkv_proj_kernel(
    const unsigned short* __restrict__ Xb,
    const unsigned short* __restrict__ Wqb, const float* __restrict__ bq,
    const unsigned short* __restrict__ Wkb, const float* __restrict__ bk,
    const unsigned short* __restrict__ Wvb, const float* __restrict__ bv,
    unsigned short* __restrict__ Qg,
    unsigned short* __restrict__ Kg,
    unsigned short* __restrict__ Vt)
{
    const int lane  = threadIdx.x & 31;
    const int wave  = threadIdx.x >> 5;
    const int l15   = lane & 15;
    const int lhalf = lane >> 4;

    const int mbase = blockIdx.x * 16;
    const int which = blockIdx.y;           // 0=Q 1=K 2=V

    const unsigned short* W = (which == 0) ? Wqb : ((which == 1) ? Wkb : Wvb);
    const float* bias       = (which == 0) ? bq  : ((which == 1) ? bk  : bv);

    const int arow = mbase + l15;

    v8f acc[6];
#pragma unroll
    for (int j = 0; j < 6; ++j) acc[j] = (v8f){0.f,0.f,0.f,0.f,0.f,0.f,0.f,0.f};

    for (int kk = 0; kk < H_DIM; kk += 32) {
        const int ka = kk + lhalf * 8;
        const uint4* ap = (const uint4*)(Xb + (size_t)arow * H_DIM + ka);
        BF16Frag A;
        A.q[0] = ap[0];
        A.q[1] = ap[2];                      // +16 elements = +32B
        BF16Frag Bf[6];
#pragma unroll
        for (int j = 0; j < 6; ++j) {
            const int brow = wave * 96 + j * 16 + l15;
            const uint4* bp = (const uint4*)(W + (size_t)brow * H_DIM + ka);
            Bf[j].q[0] = bp[0];
            Bf[j].q[1] = bp[2];
        }
#pragma unroll
        for (int j = 0; j < 6; ++j) {
            acc[j] = __builtin_amdgcn_wmma_f32_16x16x32_bf16(
                false, A.v, false, Bf[j].v, (short)0, acc[j], false, false);
        }
    }

#pragma unroll
    for (int j = 0; j < 6; ++j) {
        const int n  = wave * 96 + j * 16 + l15;
        const float bn = bias[n];
        if (which < 2) {
            unsigned short* dst = (which == 0) ? Qg : Kg;
#pragma unroll
            for (int i = 0; i < 8; ++i) {
                const int m = mbase + i + 8 * lhalf;
                dst[(size_t)m * H_DIM + n] = f32_to_bf16(acc[j][i] + bn);
            }
        } else {
            // V transposed: lane's 8 values are consecutive along s -> 1 store
            const int b  = mbase >> 12;
            const int s0 = (mbase & 4095) + 8 * lhalf;
            Pack8 pk;
#pragma unroll
            for (int i = 0; i < 8; ++i) pk.h[i] = f32_to_bf16(acc[j][i] + bn);
            *(uint4*)(Vt + (((size_t)(b * H_DIM + n)) << 12) + s0) = pk.q;
        }
    }
}

// ---------------------------------------------------------------------------
// Kernel 2: flash attention (dynamic LDS, double-buffered TDM K tiles).
// One block (8 waves / 256 thr) owns 16 query rows. Per 32-key block:
//   - wave 0 issues TENSOR_LOAD_TO_LDS for the NEXT K tile, then waits
//     s_wait_tensorcnt<=1 so the CURRENT tile is resident (in-order TDM)
//   - each wave: partial QK^T over its 96-wide k-slice from LDS (6 wmma),
//     reduced across waves with ds_add_f32
//   - wave-0 lanes run the online softmax (P bf16, corr, m, l in LDS)
//   - all waves rescale ctx, then P*V (6 wmma) with contiguous Vt loads
// ---------------------------------------------------------------------------
#define KTILE_ELEMS (32 * H_DIM)            // 24576 bf16 = 48 KB
#define FLASH_LDS_BYTES (2 * KTILE_ELEMS * 2 + 16 * 32 * 4 + 16 * 32 * 2 + 3 * 16 * 4)

__global__ __launch_bounds__(256) void flash_attn_kernel(
    const unsigned short* __restrict__ Qg,
    const unsigned short* __restrict__ Kg,
    const unsigned short* __restrict__ Vt,
    unsigned short* __restrict__ Ctx)
{
    extern __shared__ __align__(16) char smem[];
    unsigned short* Kbuf0 = (unsigned short*)smem;                 // 48 KB
    unsigned short* Kbuf1 = Kbuf0 + KTILE_ELEMS;                   // 48 KB
    float*          Sbuf  = (float*)(Kbuf1 + KTILE_ELEMS);         // 2 KB
    unsigned short* Pbuf  = (unsigned short*)(Sbuf + 16 * 32);     // 1 KB
    float*          mrow  = (float*)(Pbuf + 16 * 32);
    float*          lrow  = mrow + 16;
    float*          corr  = lrow + 16;
    unsigned short* kbuf[2] = {Kbuf0, Kbuf1};

    const int tid   = threadIdx.x;
    const int lane  = tid & 31;
    const int wave  = tid >> 5;
    const int l15   = lane & 15;
    const int lhalf = lane >> 4;

    const int mbase = blockIdx.x * 16;       // global query-row base
    const int batch = mbase >> 12;
    const int qrow  = mbase + l15;
    const unsigned short* Kbase = Kg + (size_t)(batch << 12) * H_DIM;

    // Persistent Q A-fragments for this wave's 96-wide k-slice (3 x k32).
    BF16Frag qf[3];
#pragma unroll
    for (int j = 0; j < 3; ++j) {
        const int ko = wave * 96 + j * 32 + lhalf * 8;
        const uint4* p = (const uint4*)(Qg + (size_t)qrow * H_DIM + ko);
        qf[j].q[0] = p[0];
        qf[j].q[1] = p[2];
    }

    v8f ctx[6];
#pragma unroll
    for (int j = 0; j < 6; ++j) ctx[j] = (v8f){0.f,0.f,0.f,0.f,0.f,0.f,0.f,0.f};

    if (tid < 16) { mrow[tid] = -3.0e38f; lrow[tid] = 0.f; }
    Sbuf[tid]       = 0.f;
    Sbuf[tid + 256] = 0.f;

    // Constant part of the K-tile TDM descriptor (D# group 1):
    //   data_size=2B; tensor_dim0=768, tensor_dim1=4096; tile=768x32;
    //   tensor_dim0_stride=768.
    i32x8v g1;
    g1[0] = 0x00010000;      // workgroup_mask=0 | data_size=1 (2 bytes)
    g1[1] = 0x03000000;      // tensor_dim0[15:0]=768 in bits 63:48
    g1[2] = 0x10000000;      // tensor_dim1[15:0]=4096 in bits 95:80
    g1[3] = 0x03000000;      // tile_dim0=768 in bits 127:112
    g1[4] = 32;              // tile_dim1=32 rows
    g1[5] = (int)H_DIM;      // tensor_dim0_stride=768
    g1[6] = 0;
    g1[7] = 0;
    const unsigned klds[2] = {(unsigned)(uintptr_t)Kbuf0,
                              (unsigned)(uintptr_t)Kbuf1};

    // Kick off the first K-tile DMA before entering the loop.
    if (wave == 0) tdm_load_k32(klds[0], Kbase, g1);
    __syncthreads();

    for (int kb = 0; kb < S_LEN; kb += 32) {
        const int cur = (kb >> 5) & 1;

        // ---- TDM: prefetch next tile, wait for current (in-order) ---------
        if (wave == 0) {
            if (kb + 32 < S_LEN) {
                tdm_load_k32(klds[cur ^ 1], Kbase + (size_t)(kb + 32) * H_DIM, g1);
                __builtin_amdgcn_s_wait_tensorcnt(1);
            } else {
                __builtin_amdgcn_s_wait_tensorcnt(0);
            }
        }
        // Prefetch next block's V columns into cache meanwhile.
        if (kb + 32 < S_LEN) {
            __builtin_prefetch(
                Vt + (((size_t)(batch * H_DIM + wave * 96 + l15)) << 12) + kb + 32,
                0, 0);
        }
        __syncthreads();   // current K tile visible to all waves
        const unsigned short* kcur = kbuf[cur];

        // ---- partial S = Q * K^T over this wave's k-slice (from LDS) ------
#pragma unroll
        for (int t = 0; t < 2; ++t) {
            v8f s = {0.f,0.f,0.f,0.f,0.f,0.f,0.f,0.f};
            const int krow = t * 16 + l15;               // key within block
            BF16Frag kf[3];
#pragma unroll
            for (int j = 0; j < 3; ++j) {
                const int ko = wave * 96 + j * 32 + lhalf * 8;
                const uint4* p = (const uint4*)(kcur + (size_t)krow * H_DIM + ko);
                kf[j].q[0] = p[0];
                kf[j].q[1] = p[2];
            }
#pragma unroll
            for (int j = 0; j < 3; ++j) {
                s = __builtin_amdgcn_wmma_f32_16x16x32_bf16(
                    false, qf[j].v, false, kf[j].v, (short)0, s, false, false);
            }
#pragma unroll
            for (int i = 0; i < 8; ++i) {
                const int r = i + 8 * lhalf;
                atomicAdd(&Sbuf[r * 32 + t * 16 + l15], s[i]);  // ds_add_f32
            }
        }
        __syncthreads();

        // ---- online softmax: one LDS row per lane (wave 0, lanes 0-15) ----
        if (tid < 16) {
            const float scale = 0.125f;      // 1/sqrt(64)
            float mo = mrow[tid];
            float mx = mo;
            float sv[32];
#pragma unroll
            for (int c = 0; c < 32; ++c) {
                sv[c] = Sbuf[tid * 32 + c] * scale;
                mx = fmaxf(mx, sv[c]);
            }
            const float cf = __expf(mo - mx);
            float sum = 0.f;
#pragma unroll
            for (int c = 0; c < 32; ++c) {
                const float pr = __expf(sv[c] - mx);
                sum += pr;
                Pbuf[tid * 32 + c] = f32_to_bf16(pr);
            }
            mrow[tid] = mx;
            lrow[tid] = lrow[tid] * cf + sum;
            corr[tid] = cf;
        }
        __syncthreads();

        // ---- rescale running ctx by per-row correction --------------------
        float cf8[8];
#pragma unroll
        for (int i = 0; i < 8; ++i) cf8[i] = corr[i + 8 * lhalf];
#pragma unroll
        for (int j = 0; j < 6; ++j)
#pragma unroll
            for (int i = 0; i < 8; ++i) ctx[j][i] *= cf8[i];

        // ---- P fragment from LDS (one k=32 A-operand shared by all waves) -
        BF16Frag pf;
        {
            const uint4* p = (const uint4*)(Pbuf + l15 * 32 + lhalf * 8);
            pf.q[0] = p[0];
            pf.q[1] = p[2];
        }

        // Re-zero score accumulator for next block (already consumed).
        Sbuf[tid]       = 0.f;
        Sbuf[tid + 256] = 0.f;

        // ---- ctx += P * V  (wave owns 6 x 16 columns of hidden dim) -------
        BF16Frag vf[6];
#pragma unroll
        for (int j = 0; j < 6; ++j) {
            const int h    = wave * 96 + j * 16 + l15;
            const int toff = kb + lhalf * 8;
            const uint4* p =
                (const uint4*)(Vt + (((size_t)(batch * H_DIM + h)) << 12) + toff);
            vf[j].q[0] = p[0];
            vf[j].q[1] = p[2];
        }
#pragma unroll
        for (int j = 0; j < 6; ++j) {
            ctx[j] = __builtin_amdgcn_wmma_f32_16x16x32_bf16(
                false, pf.v, false, vf[j].v, (short)0, ctx[j], false, false);
        }
        __syncthreads();
    }

    // ---- epilogue: divide by softmax denominator, store bf16 ctx ----------
    float li[8];
#pragma unroll
    for (int i = 0; i < 8; ++i) li[i] = 1.0f / lrow[i + 8 * lhalf];
#pragma unroll
    for (int j = 0; j < 6; ++j) {
        const int n = wave * 96 + j * 16 + l15;
#pragma unroll
        for (int i = 0; i < 8; ++i) {
            const int m = mbase + i + 8 * lhalf;
            Ctx[(size_t)m * H_DIM + n] = f32_to_bf16(ctx[j][i] * li[i]);
        }
    }
}

// ---------------------------------------------------------------------------
// Kernel 3: output dense (bf16 in, fp32 out).
// ---------------------------------------------------------------------------
__global__ __launch_bounds__(256) void dense_out_kernel(
    const unsigned short* __restrict__ Ctx,
    const unsigned short* __restrict__ Wdb, const float* __restrict__ bd,
    float* __restrict__ Out)
{
    const int lane  = threadIdx.x & 31;
    const int wave  = threadIdx.x >> 5;
    const int l15   = lane & 15;
    const int lhalf = lane >> 4;

    const int mbase = blockIdx.x * 16;
    const int arow  = mbase + l15;

    v8f acc[6];
#pragma unroll
    for (int j = 0; j < 6; ++j) acc[j] = (v8f){0.f,0.f,0.f,0.f,0.f,0.f,0.f,0.f};

    for (int kk = 0; kk < H_DIM; kk += 32) {
        const int ka = kk + lhalf * 8;
        const uint4* ap = (const uint4*)(Ctx + (size_t)arow * H_DIM + ka);
        BF16Frag A;
        A.q[0] = ap[0];
        A.q[1] = ap[2];
        BF16Frag Bf[6];
#pragma unroll
        for (int j = 0; j < 6; ++j) {
            const int brow = wave * 96 + j * 16 + l15;
            const uint4* bp = (const uint4*)(Wdb + (size_t)brow * H_DIM + ka);
            Bf[j].q[0] = bp[0];
            Bf[j].q[1] = bp[2];
        }
#pragma unroll
        for (int j = 0; j < 6; ++j) {
            acc[j] = __builtin_amdgcn_wmma_f32_16x16x32_bf16(
                false, A.v, false, Bf[j].v, (short)0, acc[j], false, false);
        }
    }

#pragma unroll
    for (int j = 0; j < 6; ++j) {
        const int n  = wave * 96 + j * 16 + l15;
        const float bn = bd[n];
#pragma unroll
        for (int i = 0; i < 8; ++i) {
            const int m = mbase + i + 8 * lhalf;
            Out[(size_t)m * H_DIM + n] = acc[j][i] + bn;
        }
    }
}

// ---------------------------------------------------------------------------
// Launch
// ---------------------------------------------------------------------------
extern "C" void kernel_launch(void* const* d_in, const int* in_sizes, int n_in,
                              void* d_out, int out_size, void* d_ws, size_t ws_size,
                              hipStream_t stream) {
    const float* X  = (const float*)d_in[0];
    const float* Wq = (const float*)d_in[1];
    const float* bq = (const float*)d_in[2];
    const float* Wk = (const float*)d_in[3];
    const float* bk = (const float*)d_in[4];
    const float* Wv = (const float*)d_in[5];
    const float* bv = (const float*)d_in[6];
    const float* Wd = (const float*)d_in[7];
    const float* bd = (const float*)d_in[8];
    float* Out = (float*)d_out;

    const size_t nTok  = (size_t)NTOK * H_DIM;   // 12.58M
    const size_t nW    = (size_t)H_DIM * H_DIM;  // 589824

    unsigned short* Xb  = (unsigned short*)d_ws;
    unsigned short* Wqb = Xb  + nTok;
    unsigned short* Wkb = Wqb + nW;
    unsigned short* Wvb = Wkb + nW;
    unsigned short* Wdb = Wvb + nW;
    unsigned short* Qg  = Wdb + nW;
    unsigned short* Kg  = Qg  + nTok;
    unsigned short* Vt  = Kg  + nTok;
    unsigned short* Ctx = Vt  + nTok;

    // Allow >64KB dynamic LDS for the double-buffered flash kernel.
    static bool attr_set = false;   // host-side only; deterministic
    if (!attr_set) {
        hipFuncSetAttribute(reinterpret_cast<const void*>(flash_attn_kernel),
                            hipFuncAttributeMaxDynamicSharedMemorySize,
                            FLASH_LDS_BYTES);
        attr_set = true;
    }

    // fp32 -> bf16 pre-conversion (activations + all weights)
    {
        const int n8x = (int)(nTok / 8);
        cvt_f32_bf16_kernel<<<(n8x + 255) / 256, 256, 0, stream>>>(X, Xb, n8x);
        const int n8w = (int)(nW / 8);
        cvt_f32_bf16_kernel<<<(n8w + 255) / 256, 256, 0, stream>>>(Wq, Wqb, n8w);
        cvt_f32_bf16_kernel<<<(n8w + 255) / 256, 256, 0, stream>>>(Wk, Wkb, n8w);
        cvt_f32_bf16_kernel<<<(n8w + 255) / 256, 256, 0, stream>>>(Wv, Wvb, n8w);
        cvt_f32_bf16_kernel<<<(n8w + 255) / 256, 256, 0, stream>>>(Wd, Wdb, n8w);
    }

    dim3 gProj(NTOK / 16, 3);
    qkv_proj_kernel<<<gProj, 256, 0, stream>>>(Xb, Wqb, bq, Wkb, bk, Wvb, bv,
                                               Qg, Kg, Vt);

    flash_attn_kernel<<<dim3(NTOK / 16), 256, FLASH_LDS_BYTES, stream>>>(
        Qg, Kg, Vt, Ctx);

    dense_out_kernel<<<dim3(NTOK / 16), 256, 0, stream>>>(Ctx, Wdb, bd, Out);
}
